// EdgeGNNEncoder_89532888252999
// MI455X (gfx1250) — compile-verified, matrix operand order checked by reading, not run on hardware
//
#include <hip/hip_runtime.h>

#define NN 50000
#define EE 400000
#define HH 128
#define EDIM 16
#define LL 3

typedef __attribute__((ext_vector_type(16))) __bf16 v16bf;
typedef __attribute__((ext_vector_type(8)))  float  v8f;
typedef __attribute__((ext_vector_type(4)))  unsigned int u32x4;
typedef __attribute__((ext_vector_type(8)))  int i32x8;
typedef __attribute__((ext_vector_type(4)))  int i32x4;
typedef unsigned short u16;
typedef unsigned int   u32;

#define WMMA_BF16(a, b, c) \
  __builtin_amdgcn_wmma_f32_16x16x32_bf16(false, (a), false, (b), (short)0, (c), false, false)

#if defined(__has_builtin)
#if __has_builtin(__builtin_amdgcn_tensor_load_to_lds) && \
    __has_builtin(__builtin_amdgcn_s_wait_tensorcnt)
#define HAVE_TDM 1
#endif
#endif
#ifndef HAVE_TDM
#define HAVE_TDM 0
#endif

__device__ __forceinline__ u16 f2bf(float f) {
  u32 u = __float_as_uint(f);
  u32 r = (u + 0x7FFFu + ((u >> 16) & 1u)) >> 16;  // round-to-nearest-even
  return (u16)r;
}

// A fragment (16x32 bf16, MxK): lane half selects K offset 0/8; elements 0..7 = K+0..7,
// elements 8..15 = K+16..23 -> two 16B LDS loads.
__device__ __forceinline__ v16bf load_afrag(const u16* ldsA, int stride, int kbase, int lane) {
  const u16* p = ldsA + (lane & 15) * stride + kbase + ((lane >> 4) << 3);
  v16bf v;
  ((uint4*)&v)[0] = *(const uint4*)(p);
  ((uint4*)&v)[1] = *(const uint4*)(p + 16);
  return v;
}

// B fragment pre-swizzled fragment-major: 16 contiguous bf16 per lane (32B).
__device__ __forceinline__ v16bf load_bfrag(const u16* frag, int lane) {
  const u16* p = frag + lane * 16;
  v16bf v;
  ((uint4*)&v)[0] = *(const uint4*)(p);
  ((uint4*)&v)[1] = *(const uint4*)(p + 8);
  return v;
}

// Software-pipelined K-loop: issue loads for chunk kc+1 before WMMA of chunk kc so
// ds_load_b128 of the next fragments overlaps the matrix op (partial dscnt waits).
template <int KC>
__device__ __forceinline__ v8f gemm_tile(const u16* sA, int astride, const u16* sBw,
                                         int lane, v8f acc) {
  v16bf a0 = load_afrag(sA, astride, 0, lane);
  v16bf b0 = load_bfrag(sBw, lane);
#pragma unroll
  for (int kc = 0; kc < KC; ++kc) {
    v16bf a1 = a0, b1 = b0;
    if (kc + 1 < KC) {
      a1 = load_afrag(sA, astride, (kc + 1) * 32, lane);
      b1 = load_bfrag(sBw + (kc + 1) * 4096, lane);  // 8 frags * 512 u16 per kchunk
    }
    acc = WMMA_BF16(a0, b0, acc);
    a0 = a1;
    b0 = b1;
  }
  return acc;
}

#if HAVE_TDM
// 1-row 2D TDM descriptor: contiguous global -> contiguous LDS, 8-byte elements.
// D# packing per CDNA5 ISA ch.8 (group0: count/lds_addr/global_addr/type=2;
// group1: data_size=8B, tensor_dim0=tile_dim0=elems, dims 1.. trivial).
// This toolchain exposes the 6-arg builtin: (g0, g1, g2, g3, g4, cpol).
__device__ __forceinline__ void tdm_load_1d(const void* gsrc, void* ldst, u32 nbytes) {
  u32 lds_off = (u32)(uintptr_t)ldst;  // low 32 bits of generic LDS address = LDS offset
  unsigned long long ga = (unsigned long long)(uintptr_t)gsrc;
  u32 elems = nbytes >> 3;  // 8-byte elements (<= 65535 for 16-bit tile_dim0)
  u32x4 g0;
  g0[0] = 1u;                                           // count=1 valid, no gather
  g0[1] = lds_off;                                      // lds_addr
  g0[2] = (u32)ga;                                      // global_addr[95:64]
  g0[3] = ((u32)(ga >> 32) & 0x01FFFFFFu) | (2u << 30); // global_addr hi | type=2
  i32x8 g1;
  g1[0] = (int)(3u << 16);                    // workgroup_mask=0, data_size=3 (8B)
  g1[1] = (int)((elems & 0xFFFFu) << 16);     // tensor_dim0[15:0]
  g1[2] = (int)((elems >> 16) | (1u << 16));  // tensor_dim0[31:16] | tensor_dim1=1
  g1[3] = (int)((elems & 0xFFFFu) << 16);     // tensor_dim1 hi=0 | tile_dim0=elems
  g1[4] = 1;                                  // tile_dim1=1, tile_dim2=0
  g1[5] = (int)elems;                         // tensor_dim0_stride
  g1[6] = 0;
  g1[7] = 0;
  i32x4 gz4 = {0, 0, 0, 0};
  i32x8 gz8 = {0, 0, 0, 0, 0, 0, 0, 0};
  __builtin_amdgcn_tensor_load_to_lds(g0, g1, gz4, gz4, gz8, 0);
}
#define PRELOAD_WEIGHTS(gptr, sptr, bytes)               \
  do {                                                   \
    if (threadIdx.x < 32) {                              \
      tdm_load_1d((gptr), (sptr), (bytes));              \
      __builtin_amdgcn_s_wait_tensorcnt(0);              \
    }                                                    \
    __syncthreads();                                     \
  } while (0)
#else
#define PRELOAD_WEIGHTS(gptr, sptr, bytes)                            \
  do {                                                                \
    for (u32 i = threadIdx.x; i < (bytes) / 16; i += blockDim.x)      \
      ((uint4*)(sptr))[i] = ((const uint4*)(gptr))[i];                \
    __syncthreads();                                                  \
  } while (0)
#endif

// Convert W [K][128] f32 row-major -> bf16 fragment-major [kchunk][ntile][lane][16], zero-pad K.
__global__ void wconv_kernel(const float* __restrict__ W, int K, int kchunks,
                             u16* __restrict__ dst) {
  int total = kchunks * 8 * 512;
  for (int idx = blockIdx.x * blockDim.x + threadIdx.x; idx < total;
       idx += gridDim.x * blockDim.x) {
    int e    = idx & 15;
    int lane = (idx >> 4) & 31;
    int frag = idx >> 9;
    int nt   = frag & 7;
    int kc   = frag >> 3;
    int k = kc * 32 + ((lane >> 4) << 3) + (e & 7) + ((e >> 3) << 4);
    int n = nt * 16 + (lane & 15);
    float v = (k < K) ? W[(size_t)k * HH + n] : 0.0f;
    dst[idx] = f2bf(v);
  }
}

__global__ void zero_kernel(float* __restrict__ p, long n) {
  long i  = (long)blockIdx.x * blockDim.x + threadIdx.x;
  long st = (long)gridDim.x * blockDim.x;
  for (; i < n; i += st) p[i] = 0.0f;
}

// h = x @ projW + b ; also writes bf16 shadow copy of h.
__global__ __launch_bounds__(256) void proj_kernel(
    const float* __restrict__ x, const u16* __restrict__ wf,
    const float* __restrict__ bias, float* __restrict__ h, u16* __restrict__ hbf) {
  __shared__ __attribute__((aligned(128))) u16 sW[4 * 8 * 512];
  __shared__ __attribute__((aligned(128))) u16 sA[16 * 128];
  int tid = threadIdx.x;
  PRELOAD_WEIGHTS(wf, sW, 4 * 8 * 512 * 2);
  int lane = tid & 31, wv = tid >> 5;
  int row = tid >> 4, part = tid & 15;
  for (int tile = blockIdx.x; tile < NN / 16; tile += gridDim.x) {
    const float* xr = x + (size_t)(tile * 16 + row) * HH + part * 8;
    u16* dA = sA + row * HH + part * 8;
#pragma unroll
    for (int c = 0; c < 8; ++c) dA[c] = f2bf(xr[c]);
    __syncthreads();
    v8f acc = {};
    acc = gemm_tile<4>(sA, HH, sW + wv * 512, lane, acc);
    int col = wv * 16 + (lane & 15);
    float bv = bias[col];
    int mbase = (lane >> 4) << 3;
#pragma unroll
    for (int r = 0; r < 8; ++r) {
      float v = acc[r] + bv;
      size_t o = (size_t)(tile * 16 + mbase + r) * HH + col;
      h[o] = v;
      hbf[o] = f2bf(v);
    }
    __syncthreads();
  }
}

// Fused edge-message: m = relu(relu([h_i,h_j,e] W1 + b1) W2 + b2); atomic scatter-add into agg.
// w12f = contiguous blob: W1 frags (9*8*512 u16) then W2 frags (4*8*512 u16).
__global__ __launch_bounds__(256) void edge_msg_kernel(
    const u16* __restrict__ hbf, const float* __restrict__ eattr,
    const int* __restrict__ idx_i, const int* __restrict__ idx_j,
    const u16* __restrict__ w12f, const float* __restrict__ b1,
    const float* __restrict__ b2, float* __restrict__ agg) {
  __shared__ __attribute__((aligned(128))) u16 sW12[(9 + 4) * 8 * 512];  // 106.5 KB
  __shared__ __attribute__((aligned(128))) u16 sA[16 * 288];  // K=272 padded to 288
  __shared__ __attribute__((aligned(128))) u16 sM[16 * 128];
  int tid = threadIdx.x;
  PRELOAD_WEIGHTS(w12f, sW12, (9 + 4) * 8 * 512 * 2);
  const u16* sW1 = sW12;
  const u16* sW2 = sW12 + 9 * 8 * 512;
  int lane = tid & 31, wv = tid >> 5;
  int erow = tid >> 4, part = tid & 15;
  for (int tile = blockIdx.x; tile < EE / 16; tile += gridDim.x) {
    int edge = tile * 16 + erow;
    int ni = idx_i[edge], nj = idx_j[edge];
    *(uint4*)&sA[erow * 288 + part * 8] =
        *(const uint4*)&hbf[(size_t)ni * HH + part * 8];
    *(uint4*)&sA[erow * 288 + 128 + part * 8] =
        *(const uint4*)&hbf[(size_t)nj * HH + part * 8];
    sA[erow * 288 + 256 + part] = f2bf(eattr[(size_t)edge * EDIM + part]);
    sA[erow * 288 + 272 + part] = 0;  // zero pad K 272..287
    __syncthreads();
    v8f acc = {};
    acc = gemm_tile<9>(sA, 288, sW1 + wv * 512, lane, acc);
    int col = wv * 16 + (lane & 15);
    float bv = b1[col];
    int mbase = (lane >> 4) << 3;
#pragma unroll
    for (int r = 0; r < 8; ++r) {
      float v = acc[r] + bv;
      v = v > 0.f ? v : 0.f;
      sM[(mbase + r) * HH + col] = f2bf(v);
    }
    __syncthreads();
    v8f acc2 = {};
    acc2 = gemm_tile<4>(sM, HH, sW2 + wv * 512, lane, acc2);
    float bv2 = b2[col];
#pragma unroll
    for (int r = 0; r < 8; ++r) {
      float v = acc2[r] + bv2;
      v = v > 0.f ? v : 0.f;
      int m = mbase + r;
      int dst = idx_i[tile * 16 + m];
      atomicAdd(&agg[(size_t)dst * HH + col], v);
    }
  }
}

// h = LayerNorm(relu([h,agg] W + b) + h); refreshes f32 h and bf16 shadow.
__global__ __launch_bounds__(256) void node_upd_kernel(
    float* __restrict__ h, u16* __restrict__ hbf, const float* __restrict__ agg,
    const u16* __restrict__ wf, const float* __restrict__ bias,
    const float* __restrict__ g, const float* __restrict__ bln) {
  __shared__ __attribute__((aligned(128))) u16 sW[8 * 8 * 512];  // 65.5 KB
  __shared__ __attribute__((aligned(128))) u16 sA[16 * 256];
  __shared__ __attribute__((aligned(128))) float sOut[16 * 128];
  __shared__ float sMu[16], sRs[16];
  int tid = threadIdx.x;
  PRELOAD_WEIGHTS(wf, sW, 8 * 8 * 512 * 2);
  int lane = tid & 31, wv = tid >> 5;
  int row = tid >> 4, part = tid & 15;
  for (int tile = blockIdx.x; tile < NN / 16; tile += gridDim.x) {
    int node = tile * 16 + row;
    const float* hr = h + (size_t)node * HH;
    const float* ar = agg + (size_t)node * HH;
#pragma unroll
    for (int c = 0; c < 8; ++c) {
      sA[row * 256 + part * 8 + c]       = f2bf(hr[part * 8 + c]);
      sA[row * 256 + 128 + part * 8 + c] = f2bf(ar[part * 8 + c]);
    }
    __syncthreads();
    v8f acc = {};
    acc = gemm_tile<8>(sA, 256, sW + wv * 512, lane, acc);
    int col = wv * 16 + (lane & 15);
    float bv = bias[col];
    int mbase = (lane >> 4) << 3;
#pragma unroll
    for (int r = 0; r < 8; ++r) {
      float v = acc[r] + bv;
      v = v > 0.f ? v : 0.f;
      v += h[(size_t)(tile * 16 + mbase + r) * HH + col];  // residual (f32)
      sOut[(mbase + r) * HH + col] = v;
    }
    __syncthreads();
    if (tid < 16) {
      float mu = 0.f;
      for (int c = 0; c < HH; ++c) mu += sOut[tid * HH + c];
      mu *= (1.0f / HH);
      float var = 0.f;
      for (int c = 0; c < HH; ++c) {
        float d = sOut[tid * HH + c] - mu;
        var += d * d;
      }
      var *= (1.0f / HH);
      sMu[tid] = mu;
      sRs[tid] = rsqrtf(var + 1e-5f);
    }
    __syncthreads();
    float mu = sMu[row], rs = sRs[row];
#pragma unroll
    for (int c = 0; c < 8; ++c) {
      int cc = part * 8 + c;
      float v = (sOut[row * HH + cc] - mu) * rs * g[cc] + bln[cc];
      size_t o = (size_t)node * HH + cc;
      h[o] = v;
      hbf[o] = f2bf(v);
    }
  }
}

// Final LayerNorm + global mean pool into out[128] (out pre-zeroed).
__global__ __launch_bounds__(256) void pool_kernel(
    const float* __restrict__ h, const float* __restrict__ g,
    const float* __restrict__ bln, float* __restrict__ out) {
  __shared__ float sPart[HH];
  __shared__ float sMu[16], sRs[16];
  int tid = threadIdx.x;
  int row = tid >> 4, part = tid & 15;
  if (tid < HH) sPart[tid] = 0.f;
  for (int tile = blockIdx.x; tile < NN / 16; tile += gridDim.x) {
    __syncthreads();
    if (tid < 16) {
      const float* hr = h + (size_t)(tile * 16 + tid) * HH;
      float mu = 0.f;
      for (int c = 0; c < HH; ++c) mu += hr[c];
      mu *= (1.0f / HH);
      float var = 0.f;
      for (int c = 0; c < HH; ++c) {
        float d = hr[c] - mu;
        var += d * d;
      }
      var *= (1.0f / HH);
      sMu[tid] = mu;
      sRs[tid] = rsqrtf(var + 1e-5f);
    }
    __syncthreads();
    const float* hr = h + (size_t)(tile * 16 + row) * HH;
    float mu = sMu[row], rs = sRs[row];
#pragma unroll
    for (int c = 0; c < 8; ++c) {
      int cc = part * 8 + c;
      float v = (hr[cc] - mu) * rs * g[cc] + bln[cc];
      atomicAdd(&sPart[cc], v);
    }
  }
  __syncthreads();
  if (tid < HH) atomicAdd(&out[tid], sPart[tid] * (1.0f / NN));
}

extern "C" void kernel_launch(void* const* d_in, const int* in_sizes, int n_in,
                              void* d_out, int out_size, void* d_ws, size_t ws_size,
                              hipStream_t stream) {
  (void)in_sizes; (void)n_in; (void)ws_size;
  const float* x     = (const float*)d_in[0];
  const float* eattr = (const float*)d_in[1];
  const int*   eidx  = (const int*)d_in[2];
  const float* projW = (const float*)d_in[3];
  const float* projB = (const float*)d_in[4];
  const float* msgW1 = (const float*)d_in[5];
  const float* msgB1 = (const float*)d_in[6];
  const float* msgW2 = (const float*)d_in[7];
  const float* msgB2 = (const float*)d_in[8];
  const float* updW  = (const float*)d_in[9];
  const float* updB  = (const float*)d_in[10];
  const float* lnG   = (const float*)d_in[11];
  const float* lnB   = (const float*)d_in[12];
  const float* outG  = (const float*)d_in[13];
  const float* outB  = (const float*)d_in[14];

  char* ws = (char*)d_ws;
  float* h   = (float*)(ws + 0);                 // 25,600,000 B
  u16*   hbf = (u16*)(ws + 25600000);            // 12,800,000 B
  float* agg = (float*)(ws + 38400000);          // 25,600,000 B
  u16* wfP = (u16*)(ws + 64000000);              // 32,768 B
  // Per-layer contiguous msg blob: W1 frags (73,728 B) + W2 frags (32,768 B) = 106,496 B.
  u16* wfM = (u16*)(ws + 64000000 + 32768);
  u16* wfU = (u16*)(ws + 64000000 + 32768 + 3 * 106496);  // 3 * 65,536 B

  const size_t MBLOB = 106496 / 2;  // u16 elements per msg blob
  wconv_kernel<<<64, 256, 0, stream>>>(projW, 128, 4, wfP);
  for (int l = 0; l < LL; ++l) {
    wconv_kernel<<<64, 256, 0, stream>>>(msgW1 + (size_t)l * 272 * HH, 272, 9,
                                         wfM + (size_t)l * MBLOB);
    wconv_kernel<<<64, 256, 0, stream>>>(msgW2 + (size_t)l * 128 * HH, 128, 4,
                                         wfM + (size_t)l * MBLOB + 9 * 8 * 512);
    wconv_kernel<<<64, 256, 0, stream>>>(updW + (size_t)l * 256 * HH, 256, 8,
                                         wfU + (size_t)l * 8 * 8 * 512);
  }

  proj_kernel<<<1024, 256, 0, stream>>>(x, wfP, projB, h, hbf);

  for (int l = 0; l < LL; ++l) {
    zero_kernel<<<2048, 256, 0, stream>>>(agg, (long)NN * HH);
    edge_msg_kernel<<<2048, 256, 0, stream>>>(
        hbf, eattr, eidx, eidx + EE, wfM + (size_t)l * MBLOB,
        msgB1 + (size_t)l * HH, msgB2 + (size_t)l * HH, agg);
    node_upd_kernel<<<1024, 256, 0, stream>>>(
        h, hbf, agg, wfU + (size_t)l * 8 * 8 * 512, updB + (size_t)l * HH,
        lnG + (size_t)l * HH, lnB + (size_t)l * HH);
  }

  zero_kernel<<<1, 128, 0, stream>>>((float*)d_out, out_size);
  pool_kernel<<<512, 256, 0, stream>>>(h, outG, outB, (float*)d_out);
}
